// GraphTransformerModel_7413113553207
// MI455X (gfx1250) — matrix operand order repaired and checked by reading, not compiled, not run on hardware
//
#include <hip/hip_runtime.h>
#include <hip/hip_bf16.h>

// ---------------- problem constants ----------------
#define NN   65536
#define SS   24
#define DD   64
#define HH   4
#define HD   16
#define NFF  256
#define BB   256          // NN / NFF
#define EE   1048576
#define NDTOT ((size_t)NN * DD)   // 4,194,304

typedef __attribute__((ext_vector_type(16))) _Float16 v16h;
typedef __attribute__((ext_vector_type(8)))  _Float16 v8h;
typedef __attribute__((ext_vector_type(8)))  float    v8f;

#define CAT16(lo, hi) __builtin_shufflevector(lo, hi, 0, 1, 2, 3, 4, 5, 6, 7, \
                                              8, 9, 10, 11, 12, 13, 14, 15)

// =====================================================================
// Generic WMMA GEMM:  out[M,NCOL-panel] = A16[M,K] * Bsw[K,Ncol] + bias
//  - A16: f16 row-major.  Per-lane fragment = two contiguous 8-half runs.
//  - Bsw: f16 swizzled fragment-major:
//      Bsw[((k>>5)*NCOL + n)*32 + (k&31)] = B[k][n]
//    -> a lane's 16 B elements are one contiguous 32-byte load, and with
//       NCOL compile-time all chunk offsets fold into b128 immediates.
//  - K is chunked by 64: all 12 vector loads of a chunk are issued as one
//    clause before the 8 WMMAs (overlaps next chunk's loads with math).
//  - Output: exactly one of C (f32), C16 (f16 row-major), C16T (f16
//    transposed, C16T[n*ldt + m]) is non-null.
// =====================================================================
template <int NCOL>
static __global__ void wmma_gemm_kernel(const _Float16* __restrict__ A, int lda,
                                        const _Float16* __restrict__ Bsw,
                                        const float* __restrict__ bias,
                                        float* __restrict__ C, int ldc,
                                        _Float16* __restrict__ C16, int ldc16,
                                        _Float16* __restrict__ C16T, int ldt,
                                        int M, int K, int relu)
{
    const int wid  = blockIdx.x * (blockDim.x >> 5) + (threadIdx.x >> 5);
    const int lane = threadIdx.x & 31;
    const int m0   = wid * 16;
    const int n0   = blockIdx.y * 64;
    if (m0 >= M) return;
    const int half = lane >> 4;
    const int mrow = lane & 15;

    v8f acc[4] = {v8f{}, v8f{}, v8f{}, v8f{}};
    const _Float16* ap = A + (size_t)(m0 + mrow) * lda + (half << 4);
    const _Float16* bp = Bsw + ((size_t)n0 + mrow) * 32 + (half << 4);
    constexpr int BSTEP = NCOL * 32;   // halfs per 32-wide K step

    for (int kk = 0; kk < K; kk += 64) {
        // ---- A chunk: 4 contiguous 16-byte loads ----
        v8h a0lo = *(const v8h*)(ap + kk);
        v8h a0hi = *(const v8h*)(ap + kk + 8);
        v8h a1lo = *(const v8h*)(ap + kk + 32);
        v8h a1hi = *(const v8h*)(ap + kk + 40);
        // ---- B chunk: 8 contiguous 32-byte loads, constant offsets ----
        v16h b0[4], b1[4];
        #pragma unroll
        for (int c = 0; c < 4; ++c) {
            b0[c] = *(const v16h*)(bp + c * 512);
            b1[c] = *(const v16h*)(bp + BSTEP + c * 512);
        }
        bp += 2 * BSTEP;
        __builtin_prefetch(ap + kk + 128, 0, 3);
        __builtin_prefetch(bp, 0, 3);

        v16h a0 = CAT16(a0lo, a0hi);
        v16h a1 = CAT16(a1lo, a1hi);
        #pragma unroll
        for (int c = 0; c < 4; ++c)
            acc[c] = __builtin_amdgcn_wmma_f32_16x16x32_f16(
                false, a0, false, b0[c], (short)0, acc[c], false, false);
        #pragma unroll
        for (int c = 0; c < 4; ++c)
            acc[c] = __builtin_amdgcn_wmma_f32_16x16x32_f16(
                false, a1, false, b1[c], (short)0, acc[c], false, false);
    }
    // ---- epilogue ----
    #pragma unroll
    for (int c = 0; c < 4; ++c) {
        const int n = n0 + (c << 4) + mrow;
        const float bv = bias ? bias[n] : 0.0f;
        #pragma unroll
        for (int r = 0; r < 8; ++r) {
            const int m = m0 + r + (half << 3);
            float val = acc[c][r] + bv;
            if (relu) val = fmaxf(val, 0.0f);
            if (C)    C[(size_t)m * ldc + n] = val;
            if (C16)  C16[(size_t)m * ldc16 + n] = (_Float16)val;
            if (C16T) C16T[(size_t)n * ldt + m] = (_Float16)val;
        }
    }
}

// =====================================================================
// Weight packing (f32 -> f16, fragment-swizzled)
// =====================================================================
static __global__ void pack_swz_kernel(const float* __restrict__ src,
                                       _Float16* __restrict__ dst,
                                       int Kdim, int Ncol)
{
    int i = blockIdx.x * blockDim.x + threadIdx.x;
    if (i >= Kdim * Ncol) return;
    int k = i / Ncol, n = i - k * Ncol;
    dst[((size_t)(k >> 5) * Ncol + n) * 32 + (k & 31)] = (_Float16)src[i];
}

// W [H][D][Hd] -> canonical B[k=d][c=h*16+hd], then swizzle
static __global__ void pack_headcat_swz_kernel(const float* __restrict__ W,
                                               _Float16* __restrict__ dst)
{
    int i = blockIdx.x * blockDim.x + threadIdx.x;
    if (i >= DD * DD) return;
    int d = i >> 6, c = i & 63;
    float v = W[(size_t)(c >> 4) * DD * HD + (size_t)d * HD + (c & 15)];
    dst[((size_t)(d >> 5) * DD + c) * 32 + (d & 31)] = (_Float16)v;
}

// =====================================================================
// Input projection: h = x @ Wi + bi + pe[n % 256]  (writes f32 + f16)
// =====================================================================
static __global__ void input_proj_kernel(const float* __restrict__ x,
                                         const float* __restrict__ Wi,
                                         const float* __restrict__ bi,
                                         const float* __restrict__ pe,
                                         float* __restrict__ h,
                                         _Float16* __restrict__ h16)
{
    size_t tid = (size_t)blockIdx.x * blockDim.x + threadIdx.x;
    if (tid >= NDTOT) return;
    int n = (int)(tid >> 6), d = (int)(tid & 63);
    const float* xr = x + (size_t)n * SS;
    float acc = bi[d];
    #pragma unroll
    for (int s = 0; s < SS; ++s) acc += xr[s] * Wi[s * DD + d];
    acc += pe[(size_t)(n & (NFF - 1)) * DD + d];
    h[tid] = acc;
    h16[tid] = (_Float16)acc;
}

// =====================================================================
// Graph attention edge kernels (f32; HBM/atomic bound by design)
// =====================================================================
static __global__ void graph_init_kernel(float* __restrict__ m,
                                         float* __restrict__ den,
                                         float* __restrict__ agg)
{
    size_t i = (size_t)blockIdx.x * blockDim.x + threadIdx.x;
    if (i < (size_t)NN * HH) { m[i] = -INFINITY; den[i] = 0.0f; }
    if (i < NDTOT) agg[i] = 0.0f;
}

__device__ __forceinline__ void atomic_max_f32(float* addr, float val)
{
    if (val >= 0.0f) atomicMax((int*)addr, __float_as_int(val));
    else             atomicMin((unsigned int*)addr, __float_as_uint(val));
}

static __global__ void edge_logits_kernel(const int* __restrict__ src,
                                          const int* __restrict__ dst,
                                          const float* __restrict__ q,
                                          const float* __restrict__ k,
                                          float* __restrict__ logits,
                                          float* __restrict__ mbuf)
{
    int tid = blockIdx.x * blockDim.x + threadIdx.x;
    if (tid >= EE * HH) return;
    int e = tid >> 2, hh = tid & 3;
    int sN = src[e], dN = dst[e];
    const float* qp = q + (size_t)dN * DD + hh * HD;
    const float* kp = k + (size_t)sN * DD + hh * HD;
    float acc = 0.0f;
    #pragma unroll
    for (int i = 0; i < HD; ++i) acc += qp[i] * kp[i];
    acc *= 0.25f;
    logits[tid] = acc;
    atomic_max_f32(mbuf + (size_t)dN * HH + hh, acc);
}

static __global__ void edge_exp_kernel(const int* __restrict__ dst,
                                       float* __restrict__ logits,
                                       const float* __restrict__ mbuf,
                                       float* __restrict__ den)
{
    int tid = blockIdx.x * blockDim.x + threadIdx.x;
    if (tid >= EE * HH) return;
    int e = tid >> 2, hh = tid & 3;
    int dN = dst[e];
    float ex = expf(logits[tid] - mbuf[(size_t)dN * HH + hh]);
    logits[tid] = ex;
    atomicAdd(den + (size_t)dN * HH + hh, ex);
}

static __global__ void edge_scatter_kernel(const int* __restrict__ src,
                                           const int* __restrict__ dst,
                                           const float* __restrict__ logits,
                                           const float* __restrict__ den,
                                           const float* __restrict__ v,
                                           float* __restrict__ agg)
{
    int tid = blockIdx.x * blockDim.x + threadIdx.x;
    if (tid >= EE * HH) return;
    int e = tid >> 2, hh = tid & 3;
    int sN = src[e], dN = dst[e];
    float alpha = logits[tid] / den[(size_t)dN * HH + hh];
    const float* vp = v + (size_t)sN * DD + hh * HD;
    float* ap = agg + (size_t)dN * DD + hh * HD;
    #pragma unroll
    for (int i = 0; i < HD; ++i) atomicAdd(&ap[i], alpha * vp[i]);
}

// agg16 = f16(agg + s)   (agg/s f32 are dead afterwards)
static __global__ void fuse_add_kernel(const float* __restrict__ a,
                                       const float* __restrict__ b,
                                       _Float16* __restrict__ o16, size_t n)
{
    size_t i = (size_t)blockIdx.x * blockDim.x + threadIdx.x;
    if (i < n) o16[i] = (_Float16)(a[i] + b[i]);
}

// =====================================================================
// out = LayerNorm(a + b) over D=64, one wave per row; dual f32+f16 out
// =====================================================================
static __global__ void add_ln_kernel(const float* __restrict__ a,
                                     const float* __restrict__ b,
                                     const float* __restrict__ g,
                                     const float* __restrict__ beta,
                                     float* __restrict__ out,
                                     _Float16* __restrict__ out16, int rows)
{
    int row  = blockIdx.x * (blockDim.x >> 5) + (threadIdx.x >> 5);
    int lane = threadIdx.x & 31;
    if (row >= rows) return;
    const float* ap = a + (size_t)row * DD;
    const float* bp = b + (size_t)row * DD;
    float x0 = ap[lane] + bp[lane];
    float x1 = ap[lane + 32] + bp[lane + 32];
    float s = x0 + x1;
    #pragma unroll
    for (int off = 16; off > 0; off >>= 1) s += __shfl_xor(s, off, 32);
    float mean = s * (1.0f / 64.0f);
    float d0 = x0 - mean, d1 = x1 - mean;
    float vs = d0 * d0 + d1 * d1;
    #pragma unroll
    for (int off = 16; off > 0; off >>= 1) vs += __shfl_xor(vs, off, 32);
    float inv = rsqrtf(vs * (1.0f / 64.0f) + 1e-5f);
    float y0 = d0 * inv * g[lane]      + beta[lane];
    float y1 = d1 * inv * g[lane + 32] + beta[lane + 32];
    float* op = out + (size_t)row * DD;
    op[lane] = y0;  op[lane + 32] = y1;
    _Float16* o16 = out16 + (size_t)row * DD;
    o16[lane] = (_Float16)y0;  o16[lane + 32] = (_Float16)y1;
}

// =====================================================================
// Temporal attention: one wave per (batch, head, 16-row tile).
// q,k f16 row-major; v f16 transposed [feature][row]; out f16 row-major.
// NOTE: A's K=16..31 columns are zero in every lane, so B rows 16..31 are
// don't-care -> all lanes load the same k-row unconditionally (no exec
// divergence in the score loop).
// =====================================================================
static __global__ void temporal_attn_kernel(const _Float16* __restrict__ qt16,
                                            const _Float16* __restrict__ kt16,
                                            const _Float16* __restrict__ vtT16,
                                            _Float16* __restrict__ attnO16)
{
    __shared__ float    P [16 * NFF];   // 16 KB scores
    __shared__ _Float16 Ph[16 * NFF];   // 8 KB normalized probs (f16)
    const int lane = threadIdx.x & 31;
    const int i0   = blockIdx.x * 16;
    const int hh   = blockIdx.y;
    const int bb   = blockIdx.z;
    const int half = lane >> 4;
    const int mrow = lane & 15;
    const size_t base = (size_t)bb * NFF * DD + (size_t)hh * HD;

    // ---- A fragment: q rows i0..i0+15, K=0..15 valid (K>=16 zero pad) ----
    v16h afrag = {};
    {
        const _Float16* qrow = qt16 + base + (size_t)(i0 + mrow) * DD;
        v8h lo = *(const v8h*)(qrow + (half << 3));    // K = half*8 .. +7 (<16)
        #pragma unroll
        for (int t = 0; t < 8; ++t) afrag[t] = lo[t];  // afrag[8..15] stay 0
    }
    // ---- scores: 16 tiles of 16 columns ----
    for (int jt = 0; jt < 16; ++jt) {
        // B rows 16..31 are multiplied by zero A columns: load in all lanes.
        v8h kb = *(const v8h*)(kt16 + base + (size_t)(jt * 16 + mrow) * DD + (half << 3));
        v16h bfrag = CAT16(kb, kb);
        v8f acc = {};
        acc = __builtin_amdgcn_wmma_f32_16x16x32_f16(
            false, afrag, false, bfrag, (short)0, acc, false, false);
        #pragma unroll
        for (int r = 0; r < 8; ++r)
            P[(r + (half << 3)) * NFF + jt * 16 + mrow] = acc[r] * 0.25f;
    }
    __syncthreads();
    // ---- softmax: row = mrow, lane pair (mrow, mrow+16) splits columns ----
    {
        float* row = &P[mrow * NFF];
        _Float16* rowh = &Ph[mrow * NFF];
        const int c0 = half << 7;          // 0 or 128
        float mx = -1e30f;
        for (int c = 0; c < 128; ++c) mx = fmaxf(mx, row[c0 + c]);
        mx = fmaxf(mx, __shfl_xor(mx, 16, 32));
        float ssum = 0.0f;
        for (int c = 0; c < 128; ++c) {
            float e = expf(row[c0 + c] - mx);
            row[c0 + c] = e;
            ssum += e;
        }
        ssum += __shfl_xor(ssum, 16, 32);
        float inv = 1.0f / ssum;
        for (int c = 0; c < 128; ++c)
            rowh[c0 + c] = (_Float16)(row[c0 + c] * inv);
    }
    __syncthreads();
    // ---- attn = P(16x256) @ V(256x16) ----
    v8f acc = {};
    for (int kk = 0; kk < NFF; kk += 32) {
        const _Float16* prow = &Ph[mrow * NFF + kk];
        v8h lo = *(const v8h*)(prow + (half << 4));
        v8h hi = *(const v8h*)(prow + 8 + (half << 4));
        v16h a2 = CAT16(lo, hi);
        // V^T: 16 contiguous halfs per lane
        v16h b2 = *(const v16h*)(vtT16 + (size_t)(hh * HD + mrow) * NN
                                 + (size_t)bb * NFF + kk + (half << 4));
        acc = __builtin_amdgcn_wmma_f32_16x16x32_f16(
            false, a2, false, b2, (short)0, acc, false, false);
    }
    #pragma unroll
    for (int r = 0; r < 8; ++r)
        attnO16[base + (size_t)(i0 + r + (half << 3)) * DD + mrow] = (_Float16)acc[r];
}

// =====================================================================
// Small prediction head
// =====================================================================
static __global__ void head_kernel(const float* __restrict__ a1,
                                   const float* __restrict__ Wa2, const float* __restrict__ ba2,
                                   const float* __restrict__ Wh1, const float* __restrict__ bh1,
                                   const float* __restrict__ Wh2, const float* __restrict__ bh2,
                                   float* __restrict__ out)
{
    __shared__ float s1[64], s2[32], s3[16];
    const int bb = blockIdx.x, t = threadIdx.x;
    s1[t] = a1[(size_t)bb * 64 + t];
    __syncthreads();
    if (t < 32) {
        float acc = ba2[t];
        for (int i = 0; i < 64; ++i) acc += s1[i] * Wa2[i * 32 + t];
        s2[t] = fmaxf(acc, 0.0f);
    }
    __syncthreads();
    if (t < 16) {
        float acc = bh1[t];
        for (int i = 0; i < 32; ++i) acc += s2[i] * Wh1[i * 16 + t];
        s3[t] = fmaxf(acc, 0.0f);
    }
    __syncthreads();
    if (t == 0) {
        float acc = bh2[0];
        for (int i = 0; i < 16; ++i) acc += s3[i] * Wh2[i];
        out[bb] = 1.0f / (1.0f + expf(-acc));
    }
}

// =====================================================================
// Host launcher
// =====================================================================
extern "C" void kernel_launch(void* const* d_in, const int* in_sizes, int n_in,
                              void* d_out, int out_size, void* d_ws, size_t ws_size,
                              hipStream_t stream)
{
    const float* x    = (const float*)d_in[0];
    const int*   ei   = (const int*)d_in[1];
    const float* Wi   = (const float*)d_in[2];
    const float* bi   = (const float*)d_in[3];
    const float* pe   = (const float*)d_in[4];
    const float* Wq   = (const float*)d_in[5];
    const float* Wk   = (const float*)d_in[6];
    const float* Wv   = (const float*)d_in[7];
    const float* bq   = (const float*)d_in[8];
    const float* bk   = (const float*)d_in[9];
    const float* bv   = (const float*)d_in[10];
    const float* Ws   = (const float*)d_in[11];
    const float* bs   = (const float*)d_in[12];
    const float* Wo   = (const float*)d_in[13];
    const float* bo   = (const float*)d_in[14];
    const float* glng = (const float*)d_in[15];
    const float* glnb = (const float*)d_in[16];
    const float* Wqt  = (const float*)d_in[17];
    const float* Wkt  = (const float*)d_in[18];
    const float* Wvt  = (const float*)d_in[19];
    const float* bqt  = (const float*)d_in[20];
    const float* bkt  = (const float*)d_in[21];
    const float* bvt  = (const float*)d_in[22];
    const float* Wot  = (const float*)d_in[23];
    const float* bot  = (const float*)d_in[24];
    const float* ln1g = (const float*)d_in[25];
    const float* ln1b = (const float*)d_in[26];
    const float* W1   = (const float*)d_in[27];
    const float* b1   = (const float*)d_in[28];
    const float* W2   = (const float*)d_in[29];
    const float* b2   = (const float*)d_in[30];
    const float* ln2g = (const float*)d_in[31];
    const float* ln2b = (const float*)d_in[32];
    const float* Wa1  = (const float*)d_in[33];
    const float* ba1  = (const float*)d_in[34];
    const float* Wa2  = (const float*)d_in[35];
    const float* ba2  = (const float*)d_in[36];
    const float* Wh1  = (const float*)d_in[37];
    const float* bh1  = (const float*)d_in[38];
    const float* Wh2  = (const float*)d_in[39];
    const float* bh2  = (const float*)d_in[40];
    float* out = (float*)d_out;

    const int* e_src = ei;
    const int* e_dst = ei + EE;

    // ---------------- workspace layout ----------------
    float* ws = (float*)d_ws;
    float* buf_h    = ws + 0 * NDTOT;
    float* buf_q    = ws + 1 * NDTOT;
    float* buf_k    = ws + 2 * NDTOT;
    float* buf_v    = ws + 3 * NDTOT;
    float* buf_s    = ws + 4 * NDTOT;
    float* buf_agg  = ws + 5 * NDTOT;
    float* buf_tmp  = ws + 6 * NDTOT;
    float* buf_log  = ws + 7 * NDTOT;   // E*H == N*D
    float* buf_m    = ws + 8 * NDTOT;
    float* buf_den  = buf_m + (size_t)NN * HH;
    float* buf_h2   = buf_den + (size_t)NN * HH;
    float* buf_a1   = buf_h2 + NDTOT;
    _Float16* fb    = (_Float16*)(buf_a1 + BB * DD);
    _Float16* pWq   = fb;
    _Float16* pWk   = pWq  + 4096;
    _Float16* pWv   = pWk  + 4096;
    _Float16* pWs   = pWv  + 4096;
    _Float16* pWo   = pWs  + 4096;
    _Float16* pWqt  = pWo  + 4096;
    _Float16* pWkt  = pWqt + 4096;
    _Float16* pWvt  = pWkt + 4096;
    _Float16* pWot  = pWvt + 4096;
    _Float16* pW1   = pWot + 4096;   // 64x128
    _Float16* pW2   = pW1  + 8192;   // 128x64
    _Float16* pWa1  = pW2  + 8192;   // 16384x64

    // f16 activation mirrors overlay dead f32 regions (disjoint in time):
    _Float16* h16     = (_Float16*)buf_log;
    _Float16* agg16   = (_Float16*)buf_q;
    _Float16* h2_16   = (_Float16*)buf_s;
    _Float16* qt16    = (_Float16*)buf_q;
    _Float16* kt16    = qt16 + NDTOT;
    _Float16* vtT16   = (_Float16*)buf_k;            // [64][NN] transposed
    _Float16* attnO16 = vtT16 + NDTOT;
    _Float16* ff16    = (_Float16*)buf_v;            // [NN][128]

    const dim3 B256(256);
    const dim3 gN(NN / 128, 1);

    // ---- 0. pack weights (f16, fragment-swizzled) ----
    pack_headcat_swz_kernel<<<16, B256, 0, stream>>>(Wq, pWq);
    pack_headcat_swz_kernel<<<16, B256, 0, stream>>>(Wk, pWk);
    pack_headcat_swz_kernel<<<16, B256, 0, stream>>>(Wv, pWv);
    pack_headcat_swz_kernel<<<16, B256, 0, stream>>>(Ws, pWs);
    pack_swz_kernel<<<16,   B256, 0, stream>>>(Wo,  pWo,  64, 64);
    pack_swz_kernel<<<16,   B256, 0, stream>>>(Wqt, pWqt, 64, 64);
    pack_swz_kernel<<<16,   B256, 0, stream>>>(Wkt, pWkt, 64, 64);
    pack_swz_kernel<<<16,   B256, 0, stream>>>(Wvt, pWvt, 64, 64);
    pack_swz_kernel<<<16,   B256, 0, stream>>>(Wot, pWot, 64, 64);
    pack_swz_kernel<<<32,   B256, 0, stream>>>(W1,  pW1,  64, 128);
    pack_swz_kernel<<<32,   B256, 0, stream>>>(W2,  pW2,  128, 64);
    pack_swz_kernel<<<4096, B256, 0, stream>>>(Wa1, pWa1, 16384, 64);

    // ---- 1. input projection ----
    input_proj_kernel<<<(int)(NDTOT / 256), B256, 0, stream>>>(x, Wi, bi, pe, buf_h, h16);

    // ---- 2. q/k/v/skip graph projections (f32 out for edge gather) ----
    wmma_gemm_kernel<64><<<gN, B256, 0, stream>>>(h16, DD, pWq, bq, buf_q, DD, nullptr, 0, nullptr, 0, NN, DD, 0);
    wmma_gemm_kernel<64><<<gN, B256, 0, stream>>>(h16, DD, pWk, bk, buf_k, DD, nullptr, 0, nullptr, 0, NN, DD, 0);
    wmma_gemm_kernel<64><<<gN, B256, 0, stream>>>(h16, DD, pWv, bv, buf_v, DD, nullptr, 0, nullptr, 0, NN, DD, 0);
    wmma_gemm_kernel<64><<<gN, B256, 0, stream>>>(h16, DD, pWs, bs, buf_s, DD, nullptr, 0, nullptr, 0, NN, DD, 0);

    // ---- 3. graph attention: gather-softmax-scatter ----
    graph_init_kernel<<<(int)(NDTOT / 256), B256, 0, stream>>>(buf_m, buf_den, buf_agg);
    edge_logits_kernel<<<EE * HH / 256, B256, 0, stream>>>(e_src, e_dst, buf_q, buf_k, buf_log, buf_m);
    edge_exp_kernel<<<EE * HH / 256, B256, 0, stream>>>(e_dst, buf_log, buf_m, buf_den);
    edge_scatter_kernel<<<EE * HH / 256, B256, 0, stream>>>(e_src, e_dst, buf_log, buf_den, buf_v, buf_agg);

    // ---- 4. agg16 = f16(agg + skip) ----
    fuse_add_kernel<<<(int)(NDTOT / 256), B256, 0, stream>>>(buf_agg, buf_s, agg16, NDTOT);

    // ---- 5. output projection + 6. residual LN ----
    wmma_gemm_kernel<64><<<gN, B256, 0, stream>>>(agg16, DD, pWo, bo, buf_tmp, DD, nullptr, 0, nullptr, 0, NN, DD, 0);
    add_ln_kernel<<<NN / 8, B256, 0, stream>>>(buf_tmp, buf_h, glng, glnb, buf_h2, h2_16, NN);

    // ---- 7. temporal q/k/v projections (f16 outs; v transposed) ----
    wmma_gemm_kernel<64><<<gN, B256, 0, stream>>>(h2_16, DD, pWqt, bqt, nullptr, 0, qt16, DD, nullptr, 0, NN, DD, 0);
    wmma_gemm_kernel<64><<<gN, B256, 0, stream>>>(h2_16, DD, pWkt, bkt, nullptr, 0, kt16, DD, nullptr, 0, NN, DD, 0);
    wmma_gemm_kernel<64><<<gN, B256, 0, stream>>>(h2_16, DD, pWvt, bvt, nullptr, 0, nullptr, 0, vtT16, NN, NN, DD, 0);

    // ---- 8. temporal attention ----
    temporal_attn_kernel<<<dim3(NFF / 16, HH, BB), dim3(32), 0, stream>>>(qt16, kt16, vtT16, attnO16);

    // ---- 9. attention out-proj + 10. LN1 ----
    wmma_gemm_kernel<64><<<gN, B256, 0, stream>>>(attnO16, DD, pWot, bot, buf_tmp, DD, nullptr, 0, nullptr, 0, NN, DD, 0);
    add_ln_kernel<<<NN / 8, B256, 0, stream>>>(buf_h2, buf_tmp, ln1g, ln1b, buf_h2, h2_16, NN);

    // ---- 11-13. FFN (64 -> 128 -> 64) + LN2 ----
    wmma_gemm_kernel<128><<<dim3(NN / 128, 2), B256, 0, stream>>>(h2_16, DD, pW1, b1, nullptr, 0, ff16, 128, nullptr, 0, NN, DD, 1);
    wmma_gemm_kernel<64><<<gN, B256, 0, stream>>>(ff16, 128, pW2, b2, buf_tmp, DD, nullptr, 0, nullptr, 0, NN, 128, 0);
    add_ln_kernel<<<NN / 8, B256, 0, stream>>>(buf_h2, buf_tmp, ln2g, ln2b, buf_h2, h2_16, NN);

    // ---- 14. aggregation GEMM [256,16384]@[16384,64] + 15. head ----
    wmma_gemm_kernel<64><<<dim3(2, 1), B256, 0, stream>>>(h2_16, NFF * DD, pWa1, ba1, buf_a1, DD, nullptr, 0, nullptr, 0, BB, NFF * DD, 1);
    head_kernel<<<BB, dim3(64), 0, stream>>>(buf_a1, Wa2, ba2, Wh1, bh1, Wh2, bh2, out);
}
